// ImprovedMHDPAttention_45028437131475
// MI455X (gfx1250) — matrix operand order repaired and checked by reading, not compiled, not run on hardware
//
#include <hip/hip_runtime.h>
#include <hip/hip_bf16.h>
#include <math.h>

// ---------------------------------------------------------------------------
// Problem constants (fixed by the reference): B=2, Q=K=2048, D=1024, H=16, DH=64
// ---------------------------------------------------------------------------
#define BATCH   2
#define SEQ_Q   2048
#define SEQ_K   2048
#define DMODEL  1024
#define NHEADS  16
#define DHEAD   64
#define MROWS   (BATCH * SEQ_Q)   // 4096 token rows

typedef __attribute__((ext_vector_type(16))) __bf16 bf16x16;
typedef __attribute__((ext_vector_type(8)))  __bf16 bf16x8;
typedef __attribute__((ext_vector_type(8)))  float  f32x8;

__device__ __forceinline__ f32x8 zero8() {
    f32x8 z;
#pragma unroll
    for (int i = 0; i < 8; ++i) z[i] = 0.0f;
    return z;
}

// A operand (16x32 bf16, M x K): lane l holds row m=l%16; half=l/16 selects
// K groups {8h..8h+7} (elems 0..7) and {16+8h..16+8h+7} (elems 8..15).
__device__ __forceinline__ bf16x16 load_a_frag(const __bf16* src, int ld, int lane) {
    const int m = lane & 15, half = lane >> 4;
    const __bf16* p = src + (size_t)m * ld;
    union { bf16x16 v; bf16x8 h[2]; } u;
    u.h[0] = *reinterpret_cast<const bf16x8*>(p + 8 * half);
    u.h[1] = *reinterpret_cast<const bf16x8*>(p + 16 + 8 * half);
    return u.v;
}

// B operand (32x16 bf16, K x N): lane l holds column n=l%16; half=l/16 selects
// K groups {8h..8h+7} (elems 0..7) and {16+8h..16+8h+7} (elems 8..15) of B^T row n.
__device__ __forceinline__ bf16x16 load_b_frag(const __bf16* src, int ld, int lane) {
    const int n = lane & 15, half = lane >> 4;
    const __bf16* p = src + (size_t)n * ld;
    union { bf16x16 v; bf16x8 h[2]; } u;
    u.h[0] = *reinterpret_cast<const bf16x8*>(p + 8 * half);
    u.h[1] = *reinterpret_cast<const bf16x8*>(p + 16 + 8 * half);
    return u.v;
}

__device__ __forceinline__ f32x8 wmma_bf16(bf16x16 a, bf16x16 b, f32x8 c) {
    return __builtin_amdgcn_wmma_f32_16x16x32_bf16(
        /*neg_a=*/false, a, /*neg_b=*/false, b,
        /*c_mod=*/(short)0, c, /*reuse_a=*/false, /*reuse_b=*/false);
}

// ---------------------------------------------------------------------------
// fp32 -> bf16 conversion kernels (once per launch; weights stored transposed
// so WMMA B-operand loads are K-contiguous)
// ---------------------------------------------------------------------------
__global__ void cvt_bf16_kernel(const float* __restrict__ src,
                                __bf16* __restrict__ dst, int n) {
    int i = blockIdx.x * 256 + threadIdx.x;
    if (i < n) dst[i] = (__bf16)src[i];
}

// src[R][C] fp32 -> dst[C][R] bf16
__global__ void transpose_cvt_kernel(const float* __restrict__ src,
                                     __bf16* __restrict__ dst, int R, int C) {
    int i = blockIdx.x * 256 + threadIdx.x;
    if (i < R * C) {
        int r = i / C, c = i % C;
        dst[(size_t)c * R + r] = (__bf16)src[i];
    }
}

// ---------------------------------------------------------------------------
// Fused QKV projection + RMSNorm (+ q scaling) kernel.
// grid = (MROWS/128, NHEADS, 3), block = 128 (4 waves, each wave: 32 rows x 64 cols).
// which = blockIdx.z: 0 -> q (rmsnorm*scale_q*1/8), 1 -> k (rmsnorm*scale_k),
//                     2 -> v (no norm, stored transposed [b][h][dh][token]).
// ---------------------------------------------------------------------------
__global__ __launch_bounds__(128)
void proj_qkv_kernel(const __bf16* __restrict__ xq, const __bf16* __restrict__ xkv,
                     const __bf16* __restrict__ wqt, const __bf16* __restrict__ wkt,
                     const __bf16* __restrict__ wvt,
                     const float* __restrict__ scale_q, const float* __restrict__ scale_k,
                     __bf16* __restrict__ qb, __bf16* __restrict__ kb,
                     __bf16* __restrict__ vt) {
    __shared__ __align__(16) float sT[128 * DHEAD];  // 128 rows x 64 cols (32KB)
    __shared__ float sF[128];                        // per-row rms factors

    const int which = blockIdx.z;
    const int h     = blockIdx.y;
    const int r0    = blockIdx.x * 128;
    const int tid   = threadIdx.x;
    const int wv    = tid >> 5;
    const int lane  = tid & 31;
    const int ln    = lane & 15;
    const int half  = lane >> 4;

    const __bf16* X = (which == 0) ? xq : xkv;
    const __bf16* W = (which == 0) ? wqt : ((which == 1) ? wkt : wvt);

    f32x8 acc[2][4];
#pragma unroll
    for (int mt = 0; mt < 2; ++mt)
#pragma unroll
        for (int nt = 0; nt < 4; ++nt) acc[mt][nt] = zero8();

    const __bf16* arow = X + (size_t)(r0 + wv * 32) * DMODEL;
    for (int k0 = 0; k0 < DMODEL; k0 += 32) {
        bf16x16 a0 = load_a_frag(arow + k0, DMODEL, lane);
        bf16x16 a1 = load_a_frag(arow + (size_t)16 * DMODEL + k0, DMODEL, lane);
#pragma unroll
        for (int nt = 0; nt < 4; ++nt) {
            bf16x16 b = load_b_frag(W + (size_t)(h * DHEAD + nt * 16) * DMODEL + k0,
                                    DMODEL, lane);
            acc[0][nt] = wmma_bf16(a0, b, acc[0][nt]);
            acc[1][nt] = wmma_bf16(a1, b, acc[1][nt]);
        }
    }

    if (which == 2) {
        // V: store transposed (b, h, dh, token) so P@V B-operand is K-contiguous
#pragma unroll
        for (int mt = 0; mt < 2; ++mt) {
#pragma unroll
            for (int nt = 0; nt < 4; ++nt) {
#pragma unroll
                for (int r = 0; r < 8; ++r) {
                    int row   = r0 + wv * 32 + mt * 16 + r + 8 * half;
                    int bb    = row >> 11;          // / SEQ_Q
                    int token = row & (SEQ_Q - 1);
                    int dh    = nt * 16 + ln;
                    vt[((size_t)((bb * NHEADS + h) * DHEAD + dh)) * SEQ_K + token] =
                        (__bf16)acc[mt][nt][r];
                }
            }
        }
    } else {
        // q/k: per-head RMSNorm over DH=64 columns
#pragma unroll
        for (int mt = 0; mt < 2; ++mt)
#pragma unroll
            for (int nt = 0; nt < 4; ++nt)
#pragma unroll
                for (int r = 0; r < 8; ++r)
                    sT[(wv * 32 + mt * 16 + r + 8 * half) * DHEAD + nt * 16 + ln] =
                        acc[mt][nt][r];
        __syncthreads();
        {   // every thread owns exactly one row (128 threads, 128 rows)
            const float* rowp = &sT[tid * DHEAD];
            float ss = 0.0f;
            for (int j = 0; j < DHEAD; ++j) ss += rowp[j] * rowp[j];
            float f = rsqrtf(ss * (1.0f / DHEAD) + 1e-6f);
            if (which == 0) f *= 0.125f;   // 1/sqrt(DH)
            sF[tid] = f;
        }
        __syncthreads();
        const float* sc = (which == 0) ? scale_q : scale_k;
        __bf16* dst     = (which == 0) ? qb : kb;
#pragma unroll
        for (int mt = 0; mt < 2; ++mt) {
#pragma unroll
            for (int nt = 0; nt < 4; ++nt) {
#pragma unroll
                for (int r = 0; r < 8; ++r) {
                    int br  = wv * 32 + mt * 16 + r + 8 * half;
                    int row = r0 + br;
                    int dh  = nt * 16 + ln;
                    dst[(size_t)row * DMODEL + h * DHEAD + dh] =
                        (__bf16)(acc[mt][nt][r] * sF[br] * sc[dh]);
                }
            }
        }
    }
}

// ---------------------------------------------------------------------------
// Flash attention: grid = (SEQ_Q/64, NHEADS, BATCH), block = 64 (2 waves).
// Each wave owns 32 query rows (2 x 16-row subtiles); online softmax over
// 64-key chunks; K/V fragments are reused across both subtiles.
// ---------------------------------------------------------------------------
__global__ __launch_bounds__(64)
void flash_attn_kernel(const __bf16* __restrict__ qb, const __bf16* __restrict__ kb,
                       const __bf16* __restrict__ vt, __bf16* __restrict__ xb) {
    __shared__ __align__(16) float  sS[64 * 64];   // score tile
    __shared__ __align__(16) __bf16 sP[64 * 64];   // probability tile (A layout source)
    __shared__ float sStat[64];                    // alpha / 1/l per row

    const int qblk = blockIdx.x;
    const int h    = blockIdx.y;
    const int b    = blockIdx.z;
    const int tid  = threadIdx.x;
    const int wv   = tid >> 5;
    const int lane = tid & 31;
    const int ln   = lane & 15;
    const int half = lane >> 4;
    const int qbase = qblk * 64;

    // Preload this wave's Q fragments: rows wv*32 .. wv*32+31, 2 K-steps each
    const __bf16* qptr =
        qb + (size_t)(b * SEQ_Q + qbase + wv * 32) * DMODEL + h * DHEAD;
    bf16x16 qa[2][2];
#pragma unroll
    for (int mt = 0; mt < 2; ++mt)
#pragma unroll
        for (int ks = 0; ks < 2; ++ks)
            qa[mt][ks] = load_a_frag(qptr + (size_t)(mt * 16) * DMODEL + ks * 32,
                                     DMODEL, lane);

    f32x8 o[2][4];
#pragma unroll
    for (int mt = 0; mt < 2; ++mt)
#pragma unroll
        for (int nt = 0; nt < 4; ++nt) o[mt][nt] = zero8();

    float m_i = -INFINITY, l_i = 0.0f;

    const __bf16* vhead = vt + (size_t)((b * NHEADS + h) * DHEAD) * SEQ_K;

    for (int kc = 0; kc < SEQ_K; kc += 64) {
        // Prefetch next chunk's K rows / V rows (global_prefetch_b8)
        if (kc + 64 < SEQ_K) {
            const __bf16* nk =
                kb + (size_t)(b * SEQ_K + kc + 64 + tid) * DMODEL + h * DHEAD;
            __builtin_prefetch(nk, 0, 1);
            const __bf16* nv = vhead + (size_t)tid * SEQ_K + kc + 64;
            __builtin_prefetch(nv, 0, 1);
        }

        // ---- S = Q * K^T (per wave: 32 rows x 64 keys) ----
        const __bf16* kbase = kb + (size_t)(b * SEQ_K + kc) * DMODEL + h * DHEAD;
#pragma unroll
        for (int nt = 0; nt < 4; ++nt) {
            f32x8 s0 = zero8(), s1 = zero8();
#pragma unroll
            for (int ks = 0; ks < 2; ++ks) {
                bf16x16 bk = load_b_frag(kbase + (size_t)(nt * 16) * DMODEL + ks * 32,
                                         DMODEL, lane);
                s0 = wmma_bf16(qa[0][ks], bk, s0);
                s1 = wmma_bf16(qa[1][ks], bk, s1);
            }
#pragma unroll
            for (int r = 0; r < 8; ++r) {
                sS[(wv * 32 + r + 8 * half) * 64 + nt * 16 + ln]      = s0[r];
                sS[(wv * 32 + 16 + r + 8 * half) * 64 + nt * 16 + ln] = s1[r];
            }
        }
        __syncthreads();

        // ---- online softmax: thread t owns row t (64 threads, 64 rows) ----
        {
            const float* srow = &sS[tid * 64];
            float mx = -INFINITY;
            for (int j = 0; j < 64; ++j) mx = fmaxf(mx, srow[j]);
            float mn    = fmaxf(m_i, mx);
            float alpha = __expf(m_i - mn);
            float ssum  = 0.0f;
            for (int j = 0; j < 64; ++j) {
                float p = __expf(srow[j] - mn);
                ssum += p;
                sP[tid * 64 + j] = (__bf16)p;
            }
            l_i = l_i * alpha + ssum;
            m_i = mn;
            sStat[tid] = alpha;
        }
        __syncthreads();

        // ---- rescale running O, then O += P * V ----
#pragma unroll
        for (int mt = 0; mt < 2; ++mt)
#pragma unroll
            for (int r = 0; r < 8; ++r) {
                float a = sStat[wv * 32 + mt * 16 + r + 8 * half];
#pragma unroll
                for (int nt = 0; nt < 4; ++nt) o[mt][nt][r] *= a;
            }
        bf16x16 pa[2][2];
#pragma unroll
        for (int mt = 0; mt < 2; ++mt)
#pragma unroll
            for (int ks = 0; ks < 2; ++ks)
                pa[mt][ks] = load_a_frag(&sP[(wv * 32 + mt * 16) * 64 + ks * 32],
                                         64, lane);
        const __bf16* vbase = vhead + kc;
#pragma unroll
        for (int nt = 0; nt < 4; ++nt) {
#pragma unroll
            for (int ks = 0; ks < 2; ++ks) {
                bf16x16 bv = load_b_frag(vbase + (size_t)(nt * 16) * SEQ_K + ks * 32,
                                         SEQ_K, lane);
                o[0][nt] = wmma_bf16(pa[0][ks], bv, o[0][nt]);
                o[1][nt] = wmma_bf16(pa[1][ks], bv, o[1][nt]);
            }
        }
        __syncthreads();
    }

    sStat[tid] = 1.0f / l_i;
    __syncthreads();

#pragma unroll
    for (int mt = 0; mt < 2; ++mt) {
#pragma unroll
        for (int nt = 0; nt < 4; ++nt) {
#pragma unroll
            for (int r = 0; r < 8; ++r) {
                int br    = wv * 32 + mt * 16 + r + 8 * half;
                int token = qbase + br;
                xb[(size_t)(b * SEQ_Q + token) * DMODEL + h * DHEAD + nt * 16 + ln] =
                    (__bf16)(o[mt][nt][r] * sStat[br]);
            }
        }
    }
}

// ---------------------------------------------------------------------------
// Output projection: out = x[4096,1024] @ Wo^T-layout + bo, fp32 output.
// grid = (MROWS/128, DMODEL/64), block = 128 (4 waves, 32 rows x 64 cols each).
// ---------------------------------------------------------------------------
__global__ __launch_bounds__(128)
void out_proj_kernel(const __bf16* __restrict__ xb, const __bf16* __restrict__ wot,
                     const float* __restrict__ bo, float* __restrict__ out) {
    const int r0   = blockIdx.x * 128;
    const int cb   = blockIdx.y;      // column block of 64
    const int tid  = threadIdx.x;
    const int wv   = tid >> 5;
    const int lane = tid & 31;
    const int ln   = lane & 15;
    const int half = lane >> 4;

    f32x8 acc[2][4];
#pragma unroll
    for (int mt = 0; mt < 2; ++mt)
#pragma unroll
        for (int nt = 0; nt < 4; ++nt) acc[mt][nt] = zero8();

    const __bf16* arow = xb + (size_t)(r0 + wv * 32) * DMODEL;
    for (int k0 = 0; k0 < DMODEL; k0 += 32) {
        bf16x16 a0 = load_a_frag(arow + k0, DMODEL, lane);
        bf16x16 a1 = load_a_frag(arow + (size_t)16 * DMODEL + k0, DMODEL, lane);
#pragma unroll
        for (int nt = 0; nt < 4; ++nt) {
            bf16x16 b = load_b_frag(wot + (size_t)(cb * 64 + nt * 16) * DMODEL + k0,
                                    DMODEL, lane);
            acc[0][nt] = wmma_bf16(a0, b, acc[0][nt]);
            acc[1][nt] = wmma_bf16(a1, b, acc[1][nt]);
        }
    }
#pragma unroll
    for (int mt = 0; mt < 2; ++mt) {
#pragma unroll
        for (int nt = 0; nt < 4; ++nt) {
#pragma unroll
            for (int r = 0; r < 8; ++r) {
                int row = r0 + wv * 32 + mt * 16 + r + 8 * half;
                int col = cb * 64 + nt * 16 + ln;
                out[(size_t)row * DMODEL + col] = acc[mt][nt][r] + bo[col];
            }
        }
    }
}

// ---------------------------------------------------------------------------
// Host-side launcher
// ---------------------------------------------------------------------------
extern "C" void kernel_launch(void* const* d_in, const int* in_sizes, int n_in,
                              void* d_out, int out_size, void* d_ws, size_t ws_size,
                              hipStream_t stream) {
    const float* inputs_q  = (const float*)d_in[0];  // [B,Q,D]
    const float* inputs_kv = (const float*)d_in[1];  // [B,K,D]
    const float* Wq        = (const float*)d_in[2];  // [D,H,DH]
    const float* Wk        = (const float*)d_in[3];
    const float* Wv        = (const float*)d_in[4];
    const float* scale_q   = (const float*)d_in[5];  // [DH]
    const float* scale_k   = (const float*)d_in[6];
    const float* Wo        = (const float*)d_in[7];  // [H,DH,D]
    const float* bo        = (const float*)d_in[8];  // [D]
    float* out             = (float*)d_out;

    // Workspace layout (bf16 elements)
    __bf16* p = (__bf16*)d_ws;
    __bf16* inq_bf  = p; p += (size_t)MROWS * DMODEL;         // 4M
    __bf16* inkv_bf = p; p += (size_t)MROWS * DMODEL;         // 4M
    __bf16* wq_t    = p; p += (size_t)DMODEL * DMODEL;        // 1M
    __bf16* wk_t    = p; p += (size_t)DMODEL * DMODEL;
    __bf16* wv_t    = p; p += (size_t)DMODEL * DMODEL;
    __bf16* wo_t    = p; p += (size_t)DMODEL * DMODEL;
    __bf16* q_bf    = p; p += (size_t)MROWS * DMODEL;
    __bf16* k_bf    = p; p += (size_t)MROWS * DMODEL;
    __bf16* v_t     = p; p += (size_t)MROWS * DMODEL;         // [B,H,DH,K]
    __bf16* x_bf    = p; p += (size_t)MROWS * DMODEL;

    const int nAct = MROWS * DMODEL;   // 4194304
    const int nW   = DMODEL * DMODEL;  // 1048576

    // 1) fp32 -> bf16 conversions (+ weight transposes for B-operand layout)
    cvt_bf16_kernel<<<(nAct + 255) / 256, 256, 0, stream>>>(inputs_q,  inq_bf,  nAct);
    cvt_bf16_kernel<<<(nAct + 255) / 256, 256, 0, stream>>>(inputs_kv, inkv_bf, nAct);
    transpose_cvt_kernel<<<(nW + 255) / 256, 256, 0, stream>>>(Wq, wq_t, DMODEL, DMODEL);
    transpose_cvt_kernel<<<(nW + 255) / 256, 256, 0, stream>>>(Wk, wk_t, DMODEL, DMODEL);
    transpose_cvt_kernel<<<(nW + 255) / 256, 256, 0, stream>>>(Wv, wv_t, DMODEL, DMODEL);
    transpose_cvt_kernel<<<(nW + 255) / 256, 256, 0, stream>>>(Wo, wo_t, DMODEL, DMODEL);

    // 2) fused QKV projection + RMSNorm
    {
        dim3 g(MROWS / 128, NHEADS, 3);
        proj_qkv_kernel<<<g, 128, 0, stream>>>(inq_bf, inkv_bf, wq_t, wk_t, wv_t,
                                               scale_q, scale_k, q_bf, k_bf, v_t);
    }

    // 3) flash attention
    {
        dim3 g(SEQ_Q / 64, NHEADS, BATCH);
        flash_attn_kernel<<<g, 64, 0, stream>>>(q_bf, k_bf, v_t, x_bf);
    }

    // 4) output projection + bias
    {
        dim3 g(MROWS / 128, DMODEL / 64, 1);
        out_proj_kernel<<<g, 128, 0, stream>>>(x_bf, wo_t, bo, out);
    }

    (void)in_sizes; (void)n_in; (void)out_size; (void)ws_size;
}
